// Model_72688026517883
// MI455X (gfx1250) — matrix operand order repaired and checked by reading, not compiled, not run on hardware
//
#include <hip/hip_runtime.h>
#include <math.h>

typedef __bf16 bf16;
typedef __attribute__((ext_vector_type(16))) __bf16 v16bf;
typedef __attribute__((ext_vector_type(8)))  __bf16 v8bf;
typedef __attribute__((ext_vector_type(8)))  float  v8f;

#define BATCH 1024
#define FEAT  2048
#define VOCAB 10000
#define VPAD  10048      /* vocab padded to multiple of 64 for the N-tile */
#define EMB   256
#define HID   512
#define GATESN (4*HID)
#define NDIST 3
#define SLEN  20

// ---------------------------------------------------------------------------
// GEMM: C[M,N] = A[M,K](bf16) x W[N,K](bf16)^T (+bias[N]) (+=C if accumulate)
//
// Block = 256 threads = 8 waves stacked along M. Block tile: 256(M) x 64(N).
// Wave tile: 32(M) x 64(N) = 2x4 grid of 16x16 WMMA tiles -> 8 accumulators.
// The shared 64x32 B-tile (4KB) is fetched once per block per K-step with
// GLOBAL_LOAD_ASYNC_TO_LDS_B128 (ASYNCcnt), double-buffered in LDS, and read
// back as WMMA B-fragments with ds_load_b128; the B fragment is consumed
// inline so the compiler unrolls the K-loop 2x and specializes the two LDS
// stages. A-fragments stream from global (per-wave rows, L2-resident).
// Requires M%256==0, N%64==0, K%32==0.
//
// LDS layout: row n (0..63) at byte offset n*80 (64B data + 16B pad); within-
// half-wave bank stride 20 DWORDs -> 16 distinct bank groups per phase.
// ---------------------------------------------------------------------------
__global__ void __launch_bounds__(256)
k_gemm_bf16_wmma(const bf16* __restrict__ A, const bf16* __restrict__ W,
                 float* __restrict__ C, const float* __restrict__ bias,
                 int M, int N, int K, int accumulate)
{
    __shared__ __align__(16) bf16 Bsm[2][64 * 40];   // 2 stages, row stride 40 elems (80B)

    const int tid   = threadIdx.x;
    const int lane  = tid & 31;
    const int wave  = tid >> 5;
    const int laneH = lane & 15;     // M (A) / N (B) index inside a 16x16 tile
    const int khalf = lane >> 4;     // which K-half this lane supplies

    const int n0 = blockIdx.x * 64;
    const int m0 = blockIdx.y * 256 + wave * 32;

    // Cooperative B-tile fetch: thread covers (row = tid/4, 16B segment = tid%4)
    const int brow = tid >> 2;
    const int bseg = tid & 3;
    const bf16* bsrc = W + (size_t)(n0 + brow) * K + bseg * 8;
    const unsigned blds[2] = {
        (unsigned)(uintptr_t)&Bsm[0][brow * 40 + bseg * 8],
        (unsigned)(uintptr_t)&Bsm[1][brow * 40 + bseg * 8]
    };

    const bf16* arow0 = A + (size_t)(m0 + laneH) * K + 8 * khalf;
    const bf16* arow1 = arow0 + (size_t)16 * K;

    v8f acc[2][4] = {};
    const int nk = K >> 5;

    // Prologue: stage 0 in flight
    asm volatile("global_load_async_to_lds_b128 %0, %1, off"
                 :: "v"(blds[0]), "v"(bsrc) : "memory");

    for (int kb = 0; kb < nk; ++kb) {
        const int k0 = kb << 5;

        // A fragments: independent of the LDS pipeline, overlap the wait
        __builtin_prefetch((const void*)(arow0 + k0 + 512), 0, 3);
        v8bf a0lo = *(const v8bf*)(arow0 + k0);
        v8bf a0hi = *(const v8bf*)(arow0 + k0 + 16);
        v8bf a1lo = *(const v8bf*)(arow1 + k0);
        v8bf a1hi = *(const v8bf*)(arow1 + k0 + 16);

        asm volatile("s_wait_asynccnt 0x0" ::: "memory");
        __syncthreads();                       // stage kb&1 visible to all waves
        if (kb + 1 < nk) {                     // kick off next stage into other buffer
            const bf16* nsrc = bsrc + (size_t)(kb + 1) * 32;
            asm volatile("global_load_async_to_lds_b128 %0, %1, off"
                         :: "v"(blds[(kb + 1) & 1]), "v"(nsrc) : "memory");
        }

        v16bf a0, a1;
#pragma unroll
        for (int i = 0; i < 8; ++i) {
            a0[i] = a0lo[i]; a0[i + 8] = a0hi[i];
            a1[i] = a1lo[i]; a1[i + 8] = a1hi[i];
        }

        const bf16* bs = Bsm[kb & 1];
#pragma unroll
        for (int ni = 0; ni < 4; ++ni) {
            const bf16* bp = bs + (ni * 16 + laneH) * 40 + 8 * khalf;
            v8bf blo = *(const v8bf*)(bp);
            v8bf bhi = *(const v8bf*)(bp + 16);
            v16bf b;
#pragma unroll
            for (int i = 0; i < 8; ++i) { b[i] = blo[i]; b[i + 8] = bhi[i]; }
            acc[0][ni] = __builtin_amdgcn_wmma_f32_16x16x32_bf16(
                             false, a0, false, b, (short)0, acc[0][ni], false, false);
            acc[1][ni] = __builtin_amdgcn_wmma_f32_16x16x32_bf16(
                             false, a1, false, b, (short)0, acc[1][ni], false, false);
        }
    }

    // Epilogue: C/D layout -> VGPR v holds row (v + 8*khalf), col laneH
#pragma unroll
    for (int mi = 0; mi < 2; ++mi) {
#pragma unroll
        for (int ni = 0; ni < 4; ++ni) {
            const int ncol = n0 + ni * 16 + laneH;
            const float bval = bias ? bias[ncol] : 0.0f;
#pragma unroll
            for (int v = 0; v < 8; ++v) {
                const int mrow = m0 + mi * 16 + v + 8 * khalf;
                const size_t idx = (size_t)mrow * N + ncol;
                float val = acc[mi][ni][v] + bval;
                if (accumulate) val += C[idx];
                C[idx] = val;
            }
        }
    }
}

// ---------------------------------------------------------------------------
// Elementwise / utility kernels
// ---------------------------------------------------------------------------
__global__ void k_f32_to_bf16(const float* __restrict__ in, bf16* __restrict__ out, int n) {
    int i = blockIdx.x * blockDim.x + threadIdx.x;
    if (i < n) out[i] = (bf16)in[i];
}

__global__ void k_zero_f32(float* __restrict__ p, int n) {
    int i = blockIdx.x * blockDim.x + threadIdx.x;
    if (i < n) p[i] = 0.0f;
}

__global__ void k_zero_bf16(bf16* __restrict__ p, int n) {
    int i = blockIdx.x * blockDim.x + threadIdx.x;
    if (i < n) p[i] = (bf16)0.0f;
}

__global__ void k_add_vec(const float* __restrict__ a, const float* __restrict__ b,
                          float* __restrict__ out, int n) {
    int i = blockIdx.x * blockDim.x + threadIdx.x;
    if (i < n) out[i] = a[i] + b[i];
}

__global__ void k_pad_bias(const float* __restrict__ in, float* __restrict__ out,
                           int n_valid, int n_total) {
    int i = blockIdx.x * blockDim.x + threadIdx.x;
    if (i < n_total) out[i] = (i < n_valid) ? in[i] : 0.0f;
}

__global__ void k_set_widx(int* __restrict__ widx, const int* __restrict__ tok) {
    int i = blockIdx.x * blockDim.x + threadIdx.x;
    if (i < BATCH) widx[i] = *tok;
}

// out[b,:] = bf16(emb[idx[b*stride+off], :])   (blockDim.x == EMB)
__global__ void k_gather_emb(const float* __restrict__ emb, const int* __restrict__ idx,
                             int stride, int off, bf16* __restrict__ out) {
    const int b = blockIdx.x;
    const int e = threadIdx.x;
    const int tok = idx[b * stride + off];
    out[(size_t)b * EMB + e] = (bf16)emb[(size_t)tok * EMB + e];
}

__device__ __forceinline__ float sigm(float x) { return 1.0f / (1.0f + __expf(-x)); }

// PyTorch gate order i,f,g,o; gates laid out [B, 4H]
__global__ void k_lstm_elem(const float* __restrict__ gates,
                            float* __restrict__ h, float* __restrict__ c,
                            bf16* __restrict__ h_bf) {
    const int idx = blockIdx.x * blockDim.x + threadIdx.x;   // over B*H
    const int b = idx / HID, j = idx - b * HID;
    const float* g = gates + (size_t)b * GATESN;
    const float gi = sigm(g[j]);
    const float gf = sigm(g[HID + j]);
    const float gg = tanhf(g[2 * HID + j]);
    const float go = sigm(g[3 * HID + j]);
    const float cn = gf * c[idx] + gi * gg;
    const float hn = go * tanhf(cn);
    c[idx] = cn;
    h[idx] = hn;
    h_bf[idx] = (bf16)hn;
}

// Per-row greedy argmax + online logsumexp over VOCAB. One block per row.
// logits row stride is VPAD; only the first VOCAB entries are scanned.
__global__ void k_argmax_lse(const float* __restrict__ logits,
                             int* __restrict__ widx, int* __restrict__ msg,
                             float* __restrict__ msg_f, float* __restrict__ lp,
                             int t, int store_lp) {
    __shared__ float sm[256];
    __shared__ float ss[256];
    __shared__ int   si[256];
    const int b = blockIdx.x;
    const float* row = logits + (size_t)b * VPAD;

    float m = -INFINITY, s = 0.0f;
    int mi = 0;
    for (int v = threadIdx.x; v < VOCAB; v += 256) {
        const float x = row[v];
        if (x > m) { s = s * __expf(m - x) + 1.0f; m = x; mi = v; }
        else       { s += __expf(x - m); }
    }
    sm[threadIdx.x] = m; ss[threadIdx.x] = s; si[threadIdx.x] = mi;
    __syncthreads();
    for (int off = 128; off > 0; off >>= 1) {
        if (threadIdx.x < off) {
            const float m1 = sm[threadIdx.x], m2 = sm[threadIdx.x + off];
            const float s1 = ss[threadIdx.x], s2 = ss[threadIdx.x + off];
            const int   i1 = si[threadIdx.x], i2 = si[threadIdx.x + off];
            const float mm = fmaxf(m1, m2);
            sm[threadIdx.x] = mm;
            ss[threadIdx.x] = s1 * __expf(m1 - mm) + s2 * __expf(m2 - mm);
            si[threadIdx.x] = (m1 > m2) ? i1 : (m2 > m1) ? i2 : (i1 < i2 ? i1 : i2);
        }
        __syncthreads();
    }
    if (threadIdx.x == 0) {
        widx[b] = si[0];
        msg[(size_t)b * SLEN + t] = si[0];
        msg_f[(size_t)b * SLEN + t] = (float)si[0];
        if (store_lp) lp[b] = -__logf(ss[0]);   // logit[amax] - lse == -log(sumexp)
    }
}

// loss = mean over [B,B] of (-hinge * lp[b]); hinge = sum_d relu(1 - ts + ds)
__global__ void k_loss(const float* __restrict__ ts, const float* __restrict__ ds,
                       const float* __restrict__ lp, float* __restrict__ out_loss) {
    __shared__ float red[256];
    const size_t i = (size_t)blockIdx.x * 256 + threadIdx.x;   // over B*B
    const int b = (int)(i / BATCH);
    const float t = ts[i];
    float hinge = 0.0f;
#pragma unroll
    for (int d = 0; d < NDIST; ++d) {
        const float v = 1.0f - t + ds[(size_t)d * BATCH * BATCH + i];
        hinge += v > 0.0f ? v : 0.0f;
    }
    red[threadIdx.x] = -hinge * lp[b] * (1.0f / ((float)BATCH * (float)BATCH));
    __syncthreads();
    for (int off = 128; off > 0; off >>= 1) {
        if (threadIdx.x < off) red[threadIdx.x] += red[threadIdx.x + off];
        __syncthreads();
    }
    if (threadIdx.x == 0) atomicAdd(out_loss, red[0]);
}

// accuracy: argmax([mean_n exp(ts[b,:]), mean_n exp(ds[d,b,:])...]) == 0
__global__ void k_accuracy(const float* __restrict__ ts, const float* __restrict__ ds,
                           float* __restrict__ out_acc) {
    __shared__ float red[4][256];
    const int b = blockIdx.x;
    float s0 = 0.f, s1 = 0.f, s2 = 0.f, s3 = 0.f;
    for (int n = threadIdx.x; n < BATCH; n += 256) {
        s0 += __expf(ts[(size_t)b * BATCH + n]);
        s1 += __expf(ds[(size_t)0 * BATCH * BATCH + (size_t)b * BATCH + n]);
        s2 += __expf(ds[(size_t)1 * BATCH * BATCH + (size_t)b * BATCH + n]);
        s3 += __expf(ds[(size_t)2 * BATCH * BATCH + (size_t)b * BATCH + n]);
    }
    red[0][threadIdx.x] = s0; red[1][threadIdx.x] = s1;
    red[2][threadIdx.x] = s2; red[3][threadIdx.x] = s3;
    __syncthreads();
    for (int off = 128; off > 0; off >>= 1) {
        if (threadIdx.x < off)
#pragma unroll
            for (int k = 0; k < 4; ++k) red[k][threadIdx.x] += red[k][threadIdx.x + off];
        __syncthreads();
    }
    if (threadIdx.x == 0) {
        const float tp = red[0][0];
        const int correct = (tp >= red[1][0]) && (tp >= red[2][0]) && (tp >= red[3][0]);
        atomicAdd(out_acc, correct ? (1.0f / (float)BATCH) : 0.0f);
    }
}

// ---------------------------------------------------------------------------
// Host-side orchestration
// ---------------------------------------------------------------------------
static inline void gemm(const bf16* A, const bf16* W, float* C, const float* bias,
                        int M, int N, int K, int acc, hipStream_t s) {
    dim3 grid(N / 64, M / 256);
    k_gemm_bf16_wmma<<<grid, dim3(256), 0, s>>>(A, W, C, bias, M, N, K, acc);
}

extern "C" void kernel_launch(void* const* d_in, const int* in_sizes, int n_in,
                              void* d_out, int out_size, void* d_ws, size_t ws_size,
                              hipStream_t stream) {
    (void)in_sizes; (void)n_in; (void)out_size; (void)ws_size;
    const float* target    = (const float*)d_in[0];
    const float* distr     = (const float*)d_in[1];
    const float* s_W_aff   = (const float*)d_in[2];
    const float* s_b_aff   = (const float*)d_in[3];
    const float* s_W_ih    = (const float*)d_in[4];
    const float* s_W_hh    = (const float*)d_in[5];
    const float* s_b_ih    = (const float*)d_in[6];
    const float* s_b_hh    = (const float*)d_in[7];
    const float* s_emb     = (const float*)d_in[8];
    const float* s_W_probs = (const float*)d_in[9];
    const float* s_b_probs = (const float*)d_in[10];
    const float* r_emb     = (const float*)d_in[11];
    const float* r_W_ih    = (const float*)d_in[12];
    const float* r_W_hh    = (const float*)d_in[13];
    const float* r_b_ih    = (const float*)d_in[14];
    const float* r_b_hh    = (const float*)d_in[15];
    const float* r_W_aff   = (const float*)d_in[16];
    const float* r_b_aff   = (const float*)d_in[17];
    const int*   start_tok = (const int*)d_in[18];

    float* out_loss = (float*)d_out;
    float* out_acc  = out_loss + 1;
    float* out_msg  = out_loss + 2;   // [B, SLEN] as float

    char* wsp = (char*)d_ws;
    auto take = [&](size_t bytes) -> char* {
        char* p = wsp; wsp += (bytes + 255) & ~(size_t)255; return p;
    };
    bf16*  target_bf = (bf16*)take((size_t)BATCH * FEAT * 2);
    bf16*  dist_bf   = (bf16*)take((size_t)NDIST * BATCH * FEAT * 2);
    bf16*  sWaff_bf  = (bf16*)take((size_t)HID * FEAT * 2);
    bf16*  sWih_bf   = (bf16*)take((size_t)GATESN * EMB * 2);
    bf16*  sWhh_bf   = (bf16*)take((size_t)GATESN * HID * 2);
    bf16*  sWpr_bf   = (bf16*)take((size_t)VPAD * HID * 2);     // zero-padded rows
    bf16*  rWih_bf   = (bf16*)take((size_t)GATESN * EMB * 2);
    bf16*  rWhh_bf   = (bf16*)take((size_t)GATESN * HID * 2);
    bf16*  rWaff_bf  = (bf16*)take((size_t)FEAT * HID * 2);
    float* s_bias    = (float*)take(GATESN * 4);
    float* r_bias    = (float*)take(GATESN * 4);
    float* sbpr_pad  = (float*)take((size_t)VPAD * 4);
    float* h         = (float*)take((size_t)BATCH * HID * 4);
    float* c         = (float*)take((size_t)BATCH * HID * 4);
    bf16*  h_bf      = (bf16*)take((size_t)BATCH * HID * 2);
    bf16*  w_bf      = (bf16*)take((size_t)BATCH * EMB * 2);
    float* gbuf      = (float*)take((size_t)BATCH * GATESN * 4);
    float* logits    = (float*)take((size_t)BATCH * VPAD * 4);
    int*   widx      = (int*)take(BATCH * 4);
    int*   msg       = (int*)take((size_t)BATCH * SLEN * 4);
    float* lp        = (float*)take(BATCH * 4);
    float* rt        = (float*)take((size_t)BATCH * FEAT * 4);
    bf16*  rt_bf     = (bf16*)take((size_t)BATCH * FEAT * 2);
    float* ts        = (float*)take((size_t)BATCH * BATCH * 4);
    float* ds        = (float*)take((size_t)NDIST * BATCH * BATCH * 4);

#define CONV(src, dst, n) \
    k_f32_to_bf16<<<((n) + 255) / 256, 256, 0, stream>>>(src, dst, (int)(n))

    // Weight / input conversion to bf16 (all fit in 192MB L2; once per launch)
    CONV(target,    target_bf, BATCH * FEAT);
    CONV(distr,     dist_bf,   NDIST * BATCH * FEAT);
    CONV(s_W_aff,   sWaff_bf,  HID * FEAT);
    CONV(s_W_ih,    sWih_bf,   GATESN * EMB);
    CONV(s_W_hh,    sWhh_bf,   GATESN * HID);
    CONV(s_W_probs, sWpr_bf,   VOCAB * HID);
    k_zero_bf16<<<((VPAD - VOCAB) * HID + 255) / 256, 256, 0, stream>>>(
        sWpr_bf + (size_t)VOCAB * HID, (VPAD - VOCAB) * HID);
    CONV(r_W_ih,    rWih_bf,   GATESN * EMB);
    CONV(r_W_hh,    rWhh_bf,   GATESN * HID);
    CONV(r_W_aff,   rWaff_bf,  FEAT * HID);
    k_add_vec<<<(GATESN + 255) / 256, 256, 0, stream>>>(s_b_ih, s_b_hh, s_bias, GATESN);
    k_add_vec<<<(GATESN + 255) / 256, 256, 0, stream>>>(r_b_ih, r_b_hh, r_bias, GATESN);
    k_pad_bias<<<(VPAD + 255) / 256, 256, 0, stream>>>(s_b_probs, sbpr_pad, VOCAB, VPAD);
    k_zero_f32<<<1, 32, 0, stream>>>(out_loss, 2);

    // ---- Sender init: h0 = target @ s_W_aff^T + b; c0 = 0; w0 = emb[start] ----
    gemm(target_bf, sWaff_bf, h, s_b_aff, BATCH, HID, FEAT, 0, stream);
    CONV(h, h_bf, BATCH * HID);
    k_zero_f32<<<(BATCH * HID + 255) / 256, 256, 0, stream>>>(c, BATCH * HID);
    k_set_widx<<<(BATCH + 255) / 256, 256, 0, stream>>>(widx, start_tok);
    k_gather_emb<<<BATCH, EMB, 0, stream>>>(s_emb, widx, 1, 0, w_bf);

    // ---- Sender scan (serial dependency: argmax feeds next embedding) ----
    for (int t = 0; t < SLEN; ++t) {
        gemm(w_bf, sWih_bf, gbuf, s_bias, BATCH, GATESN, EMB, 0, stream);
        gemm(h_bf, sWhh_bf, gbuf, nullptr, BATCH, GATESN, HID, 1, stream);
        k_lstm_elem<<<(BATCH * HID) / 256, 256, 0, stream>>>(gbuf, h, c, h_bf);
        gemm(h_bf, sWpr_bf, logits, sbpr_pad, BATCH, VPAD, HID, 0, stream);
        k_argmax_lse<<<BATCH, 256, 0, stream>>>(logits, widx, msg, out_msg, lp,
                                                t, t == SLEN - 1 ? 1 : 0);
        k_gather_emb<<<BATCH, EMB, 0, stream>>>(s_emb, widx, 1, 0, w_bf);
    }

    // ---- Receiver scan ----
    k_zero_f32<<<(BATCH * HID + 255) / 256, 256, 0, stream>>>(h, BATCH * HID);
    k_zero_f32<<<(BATCH * HID + 255) / 256, 256, 0, stream>>>(c, BATCH * HID);
    k_zero_bf16<<<(BATCH * HID + 255) / 256, 256, 0, stream>>>(h_bf, BATCH * HID);
    for (int t = 0; t < SLEN; ++t) {
        k_gather_emb<<<BATCH, EMB, 0, stream>>>(r_emb, msg, SLEN, t, w_bf);
        gemm(w_bf, rWih_bf, gbuf, r_bias, BATCH, GATESN, EMB, 0, stream);
        gemm(h_bf, rWhh_bf, gbuf, nullptr, BATCH, GATESN, HID, 1, stream);
        k_lstm_elem<<<(BATCH * HID) / 256, 256, 0, stream>>>(gbuf, h, c, h_bf);
    }

    // ---- r_transform and scoring GEMMs ----
    gemm(h_bf, rWaff_bf, rt, r_b_aff, BATCH, FEAT, HID, 0, stream);
    CONV(rt, rt_bf, BATCH * FEAT);
    gemm(target_bf, rt_bf, ts, nullptr, BATCH, BATCH, FEAT, 0, stream);
    for (int d = 0; d < NDIST; ++d)
        gemm(dist_bf + (size_t)d * BATCH * FEAT, rt_bf,
             ds + (size_t)d * BATCH * BATCH, nullptr, BATCH, BATCH, FEAT, 0, stream);

    // ---- Loss + accuracy reductions ----
    k_loss<<<(BATCH * BATCH) / 256, 256, 0, stream>>>(ts, ds, lp, out_loss);
    k_accuracy<<<BATCH, 256, 0, stream>>>(ts, ds, out_acc);
#undef CONV
}
